// GNNClassifier_46858093199922
// MI455X (gfx1250) — matrix operand order repaired and checked by reading, not compiled, not run on hardware
//
#include <hip/hip_runtime.h>
#include <hip/hip_bf16.h>
#include <math.h>

typedef __attribute__((ext_vector_type(16))) __bf16 bf16x16;
typedef __attribute__((ext_vector_type(8)))  __bf16 bf16x8;
typedef __attribute__((ext_vector_type(8)))  float  f32x8;

#define NODES 50000
#define EDGES 800000
#define FOUT  192      // H * HID
#define HEADS 3
#define HIDN  64
#define NGR   50

// ---------------- fill ----------------
__global__ void fill_f32(float* __restrict__ p, float v, int n) {
  int i = blockIdx.x * blockDim.x + threadIdx.x;
  if (i < n) p[i] = v;
}

// ---------------- weight prep: f32 row-major [K x 192] -> bf16 col-major [192 x K] ----
// Wt[col*K + k] = (bf16) W[k*192 + col]  -> B fragments become contiguous 32B chunks.
__global__ void transpose_w_bf16(const float* __restrict__ W, __bf16* __restrict__ Wt, int K) {
  int i = blockIdx.x * blockDim.x + threadIdx.x;
  if (i >= K * FOUT) return;
  int col = i / K;
  int k   = i - col * K;
  Wt[i] = (__bf16)W[(size_t)k * FOUT + col];
}

// ---------------- GEMM: C[M x 192] = A[M x K] @ W[K x 192] + bias (WMMA bf16) ----
// grid = (M/16, 3), block = 128 (4 waves). Wave w handles column tile blockIdx.y*4+w.
__global__ void gemm_bias_wmma(const float* __restrict__ A,
                               const __bf16* __restrict__ Wt,   // col-major bf16 [192 x K]
                               const float* __restrict__ bias,
                               float* __restrict__ C, int K) {
  __shared__ __align__(16) __bf16 Atile[16 * 192];
  const int tid   = threadIdx.x;
  const int rbase = blockIdx.x * 16;

  // cooperative load + f32->bf16 convert of the 16 x K A tile
  for (int idx = tid; idx < 16 * K; idx += 128) {
    int r = idx / K;
    int k = idx - r * K;
    Atile[r * K + k] = (__bf16)A[(size_t)(rbase + r) * K + k];
  }
  __syncthreads();

  const int lane = tid & 31;
  const int wave = tid >> 5;
  const int ct   = blockIdx.y * 4 + wave;   // 0..11
  const int col  = ct * 16 + (lane & 15);
  const int hi   = lane >> 4;               // 0: lanes 0-15, 1: lanes 16-31

  f32x8 c = {};
  for (int k0 = 0; k0 < K; k0 += 32) {
    // A fragment (contiguous k order, same as B): lane<16: K k0+0..15 ;
    // lane>=16: K k0+16..31
    const __bf16* Arow = &Atile[(lane & 15) * K + k0 + hi * 16];
    bf16x8 alo = *(const bf16x8*)(Arow);
    bf16x8 ahi = *(const bf16x8*)(Arow + 8);
    bf16x16 a = __builtin_shufflevector(alo, ahi,
                 0,1,2,3,4,5,6,7,8,9,10,11,12,13,14,15);
    // B fragment (32x16): lane holds column `col`, K rows k0+hi*16 .. +15,
    // contiguous in the transposed bf16 weights (32B aligned).
    bf16x16 b = *(const bf16x16*)(Wt + (size_t)col * K + k0 + hi * 16);

    c = __builtin_amdgcn_wmma_f32_16x16x32_bf16(false, a, false, b,
                                                (short)0, c, false, false);
  }

  const float bv     = bias[col];
  const int   rowoff = rbase + hi * 8;   // lane<16 -> rows 0..7, lane>=16 -> 8..15
  #pragma unroll
  for (int r = 0; r < 8; ++r)
    C[(size_t)(rowoff + r) * FOUT + col] = c[r] + bv;
}

// ---------------- edge kernels ----------------
__device__ __forceinline__ void atomicMaxF(float* addr, float v) {
  if (v >= 0.0f) atomicMax((int*)addr, __float_as_int(v));
  else           atomicMin((unsigned int*)addr, __float_as_uint(v));
}

__global__ void edge_logits(const int* __restrict__ src, const int* __restrict__ dst,
                            const float* __restrict__ fs, const float* __restrict__ fd,
                            const float* __restrict__ attn,
                            float* __restrict__ logits, float* __restrict__ m) {
  int e = blockIdx.x * blockDim.x + threadIdx.x;
  if (e >= EDGES) return;
  int s = src[e], d = dst[e];
  const float4* fsr = (const float4*)(fs + (size_t)s * FOUT);
  const float4* fdr = (const float4*)(fd + (size_t)d * FOUT);
  const float4* at  = (const float4*)attn;
  #pragma unroll
  for (int h = 0; h < HEADS; ++h) {
    float acc = 0.0f;
    #pragma unroll 4
    for (int q = 0; q < 16; ++q) {
      float4 a = fsr[h * 16 + q];
      float4 b = fdr[h * 16 + q];
      float4 w = at [h * 16 + q];
      float v0 = a.x + b.x; v0 = v0 > 0.0f ? v0 : 0.2f * v0;
      float v1 = a.y + b.y; v1 = v1 > 0.0f ? v1 : 0.2f * v1;
      float v2 = a.z + b.z; v2 = v2 > 0.0f ? v2 : 0.2f * v2;
      float v3 = a.w + b.w; v3 = v3 > 0.0f ? v3 : 0.2f * v3;
      acc += v0 * w.x + v1 * w.y + v2 * w.z + v3 * w.w;
    }
    logits[(size_t)e * HEADS + h] = acc;
    atomicMaxF(&m[(size_t)d * HEADS + h], acc);
  }
}

__global__ void edge_exp(const int* __restrict__ dst,
                         float* __restrict__ logits,
                         const float* __restrict__ m,
                         float* __restrict__ den) {
  int e = blockIdx.x * blockDim.x + threadIdx.x;
  if (e >= EDGES) return;
  int d = dst[e];
  #pragma unroll
  for (int h = 0; h < HEADS; ++h) {
    float ex = __expf(logits[(size_t)e * HEADS + h] - m[(size_t)d * HEADS + h]);
    logits[(size_t)e * HEADS + h] = ex;
    atomicAdd(&den[(size_t)d * HEADS + h], ex);
  }
}

__global__ void edge_aggregate(const int* __restrict__ src, const int* __restrict__ dst,
                               const float* __restrict__ logits,
                               const float* __restrict__ fs,
                               float* __restrict__ outacc) {
  int e = blockIdx.x * blockDim.x + threadIdx.x;
  if (e >= EDGES) return;
  int s = src[e], d = dst[e];
  float w0 = logits[(size_t)e * HEADS + 0];
  float w1 = logits[(size_t)e * HEADS + 1];
  float w2 = logits[(size_t)e * HEADS + 2];
  const float4* fsr = (const float4*)(fs + (size_t)s * FOUT);
  float* od = outacc + (size_t)d * FOUT;
  #pragma unroll 4
  for (int q = 0; q < 48; ++q) {
    float4 v = fsr[q];
    float w = (q < 16) ? w0 : (q < 32) ? w1 : w2;
    atomicAdd(&od[q * 4 + 0], w * v.x);
    atomicAdd(&od[q * 4 + 1], w * v.y);
    atomicAdd(&od[q * 4 + 2], w * v.z);
    atomicAdd(&od[q * 4 + 3], w * v.w);
  }
}

__global__ void normalize_nodes(const float* __restrict__ outacc,
                                const float* __restrict__ den,
                                float* __restrict__ hout) {
  int i = blockIdx.x * blockDim.x + threadIdx.x;
  if (i >= NODES * FOUT) return;
  int n = i / FOUT;
  int j = i - n * FOUT;
  hout[i] = outacc[i] / fmaxf(den[n * HEADS + (j >> 6)], 1e-9f);
}

// ---------------- pooling + MLP ----------------
__global__ void pool_accum(const float* __restrict__ h, const int* __restrict__ gid,
                           float* __restrict__ gsum) {
  int i = blockIdx.x * blockDim.x + threadIdx.x;
  if (i >= NODES * HIDN) return;
  int n = i >> 6, o = i & 63;
  float v = (h[(size_t)n * FOUT + o] + h[(size_t)n * FOUT + 64 + o] +
             h[(size_t)n * FOUT + 128 + o]) * (1.0f / 3.0f);
  atomicAdd(&gsum[gid[n] * HIDN + o], v);
}

__global__ void pool_count(const int* __restrict__ gid, float* __restrict__ gcnt) {
  int n = blockIdx.x * blockDim.x + threadIdx.x;
  if (n >= NODES) return;
  atomicAdd(&gcnt[gid[n]], 1.0f);
}

__global__ void mlp_head(const float* __restrict__ gsum, const float* __restrict__ gcnt,
                         const float* __restrict__ W1, const float* __restrict__ b1,
                         const float* __restrict__ W2, const float* __restrict__ b2,
                         const float* __restrict__ W3, const float* __restrict__ b3,
                         float* __restrict__ out) {
  __shared__ float s0[64], s1[64], s2[32];
  int g = blockIdx.x, j = threadIdx.x;
  float cnt = fmaxf(gcnt[g], 1.0f);
  s0[j] = gsum[g * 64 + j] / cnt;
  __syncthreads();
  float a1 = b1[j];
  for (int i = 0; i < 64; ++i) a1 += s0[i] * W1[i * 64 + j];
  s1[j] = a1 > 0.0f ? a1 : 0.01f * a1;
  __syncthreads();
  if (j < 32) {
    float a2 = b2[j];
    for (int i = 0; i < 64; ++i) a2 += s1[i] * W2[i * 32 + j];
    s2[j] = a2 > 0.0f ? a2 : 0.01f * a2;
  }
  __syncthreads();
  if (j < 2) {
    float a3 = b3[j];
    for (int i = 0; i < 32; ++i) a3 += s2[i] * W3[i * 2 + j];
    out[g * 2 + j] = a3 > 0.0f ? a3 : 0.01f * a3;
  }
}

// ---------------- host launcher ----------------
extern "C" void kernel_launch(void* const* d_in, const int* in_sizes, int n_in,
                              void* d_out, int out_size, void* d_ws, size_t ws_size,
                              hipStream_t stream) {
  const float* x    = (const float*)d_in[0];
  const int*   src  = (const int*)d_in[1];
  const int*   dst  = (const int*)d_in[2];
  const int*   gid  = (const int*)d_in[3];

  const float* Wsrc[3] = { (const float*)d_in[5],  (const float*)d_in[10], (const float*)d_in[15] };
  const float* bsrc[3] = { (const float*)d_in[6],  (const float*)d_in[11], (const float*)d_in[16] };
  const float* Wdst[3] = { (const float*)d_in[7],  (const float*)d_in[12], (const float*)d_in[17] };
  const float* bdst[3] = { (const float*)d_in[8],  (const float*)d_in[13], (const float*)d_in[18] };
  const float* attn[3] = { (const float*)d_in[9],  (const float*)d_in[14], (const float*)d_in[19] };
  const int    Kin[3]  = { 128, FOUT, FOUT };

  const float* W1 = (const float*)d_in[20]; const float* b1 = (const float*)d_in[21];
  const float* W2 = (const float*)d_in[22]; const float* b2 = (const float*)d_in[23];
  const float* W3 = (const float*)d_in[24]; const float* b3 = (const float*)d_in[25];

  // workspace layout (floats, then bf16 weight staging)
  float* ws     = (float*)d_ws;
  float* hbuf   = ws;                                  // NODES*FOUT
  float* fs     = hbuf   + (size_t)NODES * FOUT;       // NODES*FOUT
  float* fdo    = fs     + (size_t)NODES * FOUT;       // fd, then reused as outacc
  float* logits = fdo    + (size_t)NODES * FOUT;       // EDGES*HEADS
  float* mbuf   = logits + (size_t)EDGES * HEADS;      // NODES*HEADS
  float* den    = mbuf   + (size_t)NODES * HEADS;      // NODES*HEADS
  float* gsum   = den    + (size_t)NODES * HEADS;      // NGR*HIDN
  float* gcnt   = gsum   + (size_t)NGR * HIDN;         // NGR
  size_t foff   = (size_t)(gcnt + NGR - ws);
  foff          = (foff + 15) & ~(size_t)15;           // 64B-align the bf16 area
  __bf16* wtS   = (__bf16*)(ws + foff);                // 192*192 bf16 (max)
  __bf16* wtD   = wtS + (size_t)FOUT * FOUT;           // 192*192 bf16 (max)

  const int eb  = (EDGES + 255) / 256;
  const int nfb = (NODES * FOUT + 255) / 256;
  const int nhb = (NODES * HEADS + 255) / 256;
  dim3 ggrid(NODES / 16, 3);

  for (int l = 0; l < 3; ++l) {
    const float* A = (l == 0) ? x : hbuf;
    int K  = Kin[l];
    int we = K * FOUT;
    transpose_w_bf16<<<(we + 255) / 256, 256, 0, stream>>>(Wsrc[l], wtS, K);
    transpose_w_bf16<<<(we + 255) / 256, 256, 0, stream>>>(Wdst[l], wtD, K);

    gemm_bias_wmma<<<ggrid, 128, 0, stream>>>(A, wtS, bsrc[l], fs,  K);
    gemm_bias_wmma<<<ggrid, 128, 0, stream>>>(A, wtD, bdst[l], fdo, K);

    fill_f32<<<nhb, 256, 0, stream>>>(mbuf, -INFINITY, NODES * HEADS);
    edge_logits<<<eb, 256, 0, stream>>>(src, dst, fs, fdo, attn[l], logits, mbuf);

    fill_f32<<<nhb, 256, 0, stream>>>(den, 0.0f, NODES * HEADS);
    edge_exp<<<eb, 256, 0, stream>>>(dst, logits, mbuf, den);

    fill_f32<<<nfb, 256, 0, stream>>>(fdo, 0.0f, NODES * FOUT);  // now outacc
    edge_aggregate<<<eb, 256, 0, stream>>>(src, dst, logits, fs, fdo);

    normalize_nodes<<<nfb, 256, 0, stream>>>(fdo, den, hbuf);
  }

  fill_f32<<<(NGR * HIDN + NGR + 255) / 256, 256, 0, stream>>>(gsum, 0.0f, NGR * HIDN + NGR);
  pool_accum<<<(NODES * HIDN + 255) / 256, 256, 0, stream>>>(hbuf, gid, gsum);
  pool_count<<<(NODES + 255) / 256, 256, 0, stream>>>(gid, gcnt);
  mlp_head<<<NGR, 64, 0, stream>>>(gsum, gcnt, W1, b1, W2, b2, W3, b3, (float*)d_out);
}
